// ScannedMultiLayerGRU_65438121722410
// MI455X (gfx1250) — matrix operand order repaired and checked by reading, not compiled, not run on hardware
//
#include <hip/hip_runtime.h>

// Problem constants (reference: T=1024, B=64, W=512, L=2)
#define T_STEPS 1024
#define BATCH   64
#define WIDTH   512
#define W3      1536
#define LAYERS  2

typedef __attribute__((ext_vector_type(16))) __bf16 v16bf;
typedef __attribute__((ext_vector_type(8)))  float  v8f;

// Round-to-nearest-even f32 -> bf16
__device__ __forceinline__ unsigned short f2bf(float f) {
    unsigned u = __float_as_uint(f);
    u += 0x7FFFu + ((u >> 16) & 1u);
    return (unsigned short)(u >> 16);
}
__device__ __forceinline__ float sigmoidf_(float x) { return 1.0f / (1.0f + __expf(-x)); }

// A-pack (bf16 16x32 WMMA A layout, ISA 7.12.2):
//   lane<16 : row = mt*16+lane,    halves 0..7 = K kb*32+0..7,  halves 8..15 = K kb*32+16..23
//   lane>=16: row = mt*16+lane-16, halves 0..7 = K kb*32+8..15, halves 8..15 = K kb*32+24..31
__device__ __forceinline__ size_t apack_idx(int mt, int kb, int lane, int half) {
    return ((size_t)((mt * 16 + kb) * 32 + lane)) * 16 + half;
}

// ---------------------------------------------------------------------------
// Pack fp32 activations [rows x WIDTH] into bf16 WMMA A layout.
// One thread writes one 8-half contiguous group (reads 8 contiguous f32).
__global__ void pack_a_f32_to_bf16(const float* __restrict__ src,
                                   unsigned short* __restrict__ dst, int mtiles) {
    int tid = blockIdx.x * blockDim.x + threadIdx.x;
    int total = mtiles * 16 * 32 * 2;
    if (tid >= total) return;
    int hg   = tid & 1;
    int lane = (tid >> 1) & 31;
    int kb   = (tid >> 6) & 15;
    int mt   = tid >> 10;
    int row  = mt * 16 + (lane & 15);
    int k0   = kb * 32 + ((lane >= 16) ? 8 : 0) + hg * 16;
    const float* s = src + (size_t)row * WIDTH + k0;
    unsigned short* d = dst + apack_idx(mt, kb, lane, hg * 8);
#pragma unroll
    for (int j = 0; j < 8; ++j) d[j] = f2bf(s[j]);
}

// Pack fp32 weights [WIDTH x W3] (row=k, col=n) into bf16 WMMA B layout:
//   tile nt (96 N-tiles), kb (16 K-blocks); lane%16 = N, lane/16 selects K half:
//   lanes 0-15 hold K kb*32+0..7,16..23, lanes 16-31 hold K kb*32+8..15,24..31.
__global__ void pack_b_f32_to_bf16(const float* __restrict__ src,
                                   unsigned short* __restrict__ dst) {
    int tid = blockIdx.x * blockDim.x + threadIdx.x;
    if (tid >= 96 * 16 * 32) return;
    int lane = tid & 31;
    int kb   = (tid >> 5) & 15;
    int nt   = tid >> 9;
    int col  = nt * 16 + (lane & 15);
    int k0   = kb * 32 + ((lane >= 16) ? 8 : 0);
    unsigned short* d = dst + ((size_t)((nt * 16 + kb) * 32 + lane)) * 16;
#pragma unroll
    for (int j = 0; j < 16; ++j) d[j] = f2bf(src[(size_t)(k0 + j + (j >= 8 ? 8 : 0)) * W3 + col]);
}

// ---------------------------------------------------------------------------
// Persistent per-layer GRU scan. Grid = 32 blocks (one per 16-column tile of W),
// 4 waves/block (one per 16-row M tile of B=64). Device-wide barrier per step.
__global__ void __launch_bounds__(128) gru_scan(
    const unsigned short* __restrict__ xA_base,  // A-pack of this layer's inputs (T*4 mtiles)
    unsigned short* __restrict__ yA_base,        // layer0: A-pack output buffer; layer1: null
    const unsigned short* __restrict__ WiP,      // B-pack Wi (96 tiles x 16 kb)
    const unsigned short* __restrict__ WhP,      // B-pack Wh
    const float* __restrict__ bi,                // [3W]
    const float* __restrict__ bhn,               // [W]
    const unsigned char* __restrict__ resets,    // [T*B] bool
    unsigned short* __restrict__ hA,             // 2 x 32768 halves (double buffered)
    float* __restrict__ hF,                      // 2 x 32768 floats (double buffered)
    float* __restrict__ ys_out,                  // layer1: d_out + L*B*W; layer0: null
    float* __restrict__ carry_out,               // d_out + layer*B*W
    unsigned* __restrict__ bar)                  // grid barrier counter (zeroed)
{
    const int nt    = blockIdx.x;        // column tile 0..31
    const int wave  = threadIdx.x >> 5;  // row tile 0..3
    const int lane  = threadIdx.x & 31;
    const int laneN = lane & 15;
    const bool hi   = lane >= 16;
    const int arow  = wave * 16 + laneN;  // row owned by this lane in the A operand
    const int col   = nt * 16 + laneN;    // output column owned in C layout

    const float br = bi[col], bz = bi[WIDTH + col], bn = bi[2 * WIDTH + col];
    const float bh = bhn[col];

    // precompute A-pack (lane,half) coordinates for writing column `col`
    const int kb2 = col >> 5;
    const int c32 = col & 31;
    int lane2add, halfbase;
    if (c32 < 8)       { lane2add = 0;  halfbase = c32; }
    else if (c32 < 16) { lane2add = 16; halfbase = c32 - 8; }
    else if (c32 < 24) { lane2add = 0;  halfbase = 8 + (c32 - 16); }
    else               { lane2add = 16; halfbase = 8 + (c32 - 24); }

    for (int t = 0; t < T_STEPS; ++t) {
        const unsigned short* hin  = hA + (size_t)(t & 1) * 32768;
        unsigned short*       hout = hA + (size_t)((t + 1) & 1) * 32768;
        const float*          hfin = hF + (size_t)(t & 1) * 32768;
        float*                hfout = hF + (size_t)((t + 1) & 1) * 32768;
        const unsigned char*  rst  = resets + (size_t)t * BATCH;
        const unsigned short* xA   = xA_base + (size_t)t * 32768;

        const bool rstA = rst[arow] != 0;

        v8f accR = {}, accZ = {}, accIN = {}, accHN = {};
#pragma unroll 4
        for (int kb = 0; kb < 16; ++kb) {
            v16bf xv = *(const v16bf*)(xA + apack_idx(wave, kb, lane, 0));
            v16bf hv = *(const v16bf*)(hin + apack_idx(wave, kb, lane, 0));
            if (rstA) { v16bf zv = {}; hv = zv; }  // reset: zero hidden contribution

            const size_t wb = ((size_t)((nt * 16 + kb) * 32 + lane)) * 16;
            // gate g occupies N-tiles g*32 + nt -> +g*262144 halves
            v16bf wir = *(const v16bf*)(WiP + wb);
            v16bf wiz = *(const v16bf*)(WiP + wb + 262144);
            v16bf win = *(const v16bf*)(WiP + wb + 524288);
            v16bf whr = *(const v16bf*)(WhP + wb);
            v16bf whz = *(const v16bf*)(WhP + wb + 262144);
            v16bf whn = *(const v16bf*)(WhP + wb + 524288);

            // r/z gates: x@Wi and h@Wh share one accumulator (they are summed anyway)
            accR  = __builtin_amdgcn_wmma_f32_16x16x32_bf16(false, xv, false, wir, (short)0, accR,  false, false);
            accR  = __builtin_amdgcn_wmma_f32_16x16x32_bf16(false, hv, false, whr, (short)0, accR,  false, false);
            accZ  = __builtin_amdgcn_wmma_f32_16x16x32_bf16(false, xv, false, wiz, (short)0, accZ,  false, false);
            accZ  = __builtin_amdgcn_wmma_f32_16x16x32_bf16(false, hv, false, whz, (short)0, accZ,  false, false);
            // n gate: keep input/hidden parts separate (hn is scaled by r)
            accIN = __builtin_amdgcn_wmma_f32_16x16x32_bf16(false, xv, false, win, (short)0, accIN, false, false);
            accHN = __builtin_amdgcn_wmma_f32_16x16x32_bf16(false, hv, false, whn, (short)0, accHN, false, false);
        }

        // C/D layout: lane<16 -> M = wave*16 + v ; lane>=16 -> M = wave*16 + v + 8
#pragma unroll
        for (int v = 0; v < 8; ++v) {
            int row = wave * 16 + v + (hi ? 8 : 0);
            float hprev = rst[row] ? 0.0f : hfin[(size_t)row * WIDTH + col];
            float r = sigmoidf_(accR[v] + br);
            float z = sigmoidf_(accZ[v] + bz);
            float n = tanhf(accIN[v] + bn + r * (accHN[v] + bh));
            float hnew = (1.0f - z) * n + z * hprev;

            hfout[(size_t)row * WIDTH + col] = hnew;

            unsigned short hb = f2bf(hnew);
            int mt2 = row >> 4, r16 = row & 15;
            size_t pidx = ((size_t)((mt2 * 16 + kb2) * 32 + (r16 + lane2add))) * 16 + halfbase;
            hout[pidx] = hb;                                         // h A-pack for next step
            if (yA_base) yA_base[(size_t)t * 32768 + pidx] = hb;     // layer0 output -> layer1 input pack
            if (ys_out)  ys_out[(size_t)t * BATCH * WIDTH + (size_t)row * WIDTH + col] = hnew;
            if (t == T_STEPS - 1) carry_out[(size_t)row * WIDTH + col] = hnew;
        }

        // ---- device-wide barrier (release writes, acquire before next step's reads)
        __syncthreads();
        if (threadIdx.x == 0) {
            __hip_atomic_fetch_add(bar, 1u, __ATOMIC_RELEASE, __HIP_MEMORY_SCOPE_AGENT);
            unsigned tgt = (unsigned)(32 * (t + 1));
            while (__hip_atomic_load(bar, __ATOMIC_ACQUIRE, __HIP_MEMORY_SCOPE_AGENT) < tgt)
                __builtin_amdgcn_s_sleep(1);
        }
        __syncthreads();
    }
}

// ---------------------------------------------------------------------------
extern "C" void kernel_launch(void* const* d_in, const int* in_sizes, int n_in,
                              void* d_out, int out_size, void* d_ws, size_t ws_size,
                              hipStream_t stream) {
    (void)in_sizes; (void)n_in; (void)out_size; (void)ws_size;
    const float*         ins    = (const float*)d_in[0];          // [T,B,W] f32
    const unsigned char* resets = (const unsigned char*)d_in[1];  // [T,B] bool
    const float*         Wi     = (const float*)d_in[2];          // [L,W,3W]
    const float*         bi     = (const float*)d_in[3];          // [L,3W]
    const float*         Wh     = (const float*)d_in[4];          // [L,W,3W]
    const float*         bhn    = (const float*)d_in[5];          // [L,W]
    float* out = (float*)d_out;  // [L*B*W carry][T*B*W ys]

    // Workspace layout (bytes)
    char* ws = (char*)d_ws;
    unsigned short* ApackX = (unsigned short*)(ws);                // 64 MiB: ins bf16 A-pack
    unsigned short* ApackY = (unsigned short*)(ws + 67108864);     // 64 MiB: layer0 outputs A-pack
    unsigned short* WiP    = (unsigned short*)(ws + 134217728);    // 2 x 1.5 MiB
    unsigned short* WhP    = (unsigned short*)(ws + 137363456);    // 2 x 1.5 MiB
    unsigned short* hA     = (unsigned short*)(ws + 140509184);    // 2 x 64 KiB (double buffer)
    float*          hF     = (float*)(ws + 140640256);             // 2 x 128 KiB (double buffer)
    unsigned*       bar    = (unsigned*)(ws + 140902400);          // barrier counter

    // Pack inputs and weights to WMMA-native bf16 layouts (parallel, once per call)
    pack_a_f32_to_bf16<<<(4096 * 1024) / 256, 256, 0, stream>>>(ins, ApackX, 4096);
    for (int l = 0; l < LAYERS; ++l) {
        pack_b_f32_to_bf16<<<(96 * 16 * 32) / 256, 256, 0, stream>>>(
            Wi + (size_t)l * WIDTH * W3, WiP + (size_t)l * 786432);
        pack_b_f32_to_bf16<<<(96 * 16 * 32) / 256, 256, 0, stream>>>(
            Wh + (size_t)l * WIDTH * W3, WhP + (size_t)l * 786432);
    }

    for (int l = 0; l < LAYERS; ++l) {
        hipMemsetAsync(hA, 0, 65536, stream);    // zero h bf16 buffer0 (t=0 reads it)
        hipMemsetAsync(hF, 0, 131072, stream);   // zero h f32 buffer0
        hipMemsetAsync(bar, 0, 4, stream);       // reset barrier
        gru_scan<<<32, 128, 0, stream>>>(
            (l == 0) ? ApackX : ApackY,
            (l == 0) ? ApackY : (unsigned short*)nullptr,
            WiP + (size_t)l * 786432,
            WhP + (size_t)l * 786432,
            bi + (size_t)l * W3,
            bhn + (size_t)l * WIDTH,
            resets,
            hA, hF,
            (l == 1) ? out + (size_t)LAYERS * BATCH * WIDTH : (float*)nullptr,
            out + (size_t)l * BATCH * WIDTH,
            bar);
    }
}
